// GAT_NFM_7928509629244
// MI455X (gfx1250) — matrix operand order repaired and checked by reading, not compiled, not run on hardware
//
#include <hip/hip_runtime.h>
#include <hip/hip_bf16.h>

typedef __attribute__((ext_vector_type(16))) _Float16 v16h;
typedef __attribute__((ext_vector_type(8)))  _Float16 v8h;
typedef __attribute__((ext_vector_type(8)))  float    v8f;

struct Frag { union { v16h v; v8h h[2]; }; };

// ---------------- prep kernels ----------------
__global__ void k_cvt_f16(const float* __restrict__ in, _Float16* __restrict__ out, int n) {
  int i = blockIdx.x * blockDim.x + threadIdx.x;
  if (i < n) out[i] = (_Float16)in[i];
}

// W is [K, Nout] row-major; produce Wt [Nout, K] f16 (column-major of W).
__global__ void k_transpose_f16(const float* __restrict__ W, _Float16* __restrict__ Wt,
                                int K, int Nout, int square) {
  int i = blockIdx.x * blockDim.x + threadIdx.x;
  if (i < K * Nout) {
    int k = i / Nout, n = i - k * Nout;
    float v = W[i];
    if (square) v *= v;
    Wt[(size_t)n * K + k] = (_Float16)v;
  }
}

// ---------------- WMMA GEMM: C[M,Nout] = A[M,K] * B[K,Nout], Bt is [Nout,K] f16
// One wave computes a 16x64 tile (4 WMMA n-tiles). Nout must be a multiple of 64.
__global__ __launch_bounds__(256) void k_gemm_wmma(
    const _Float16* __restrict__ A, const _Float16* __restrict__ Bt,
    float* __restrict__ C, int K, int Nout, int totalWaves) {
  int w    = (blockIdx.x * blockDim.x + threadIdx.x) >> 5;
  int lane = threadIdx.x & 31;
  if (w >= totalWaves) return;
  int ntiles4 = Nout >> 6;
  int mt = w / ntiles4;
  int nt = (w - mt * ntiles4) << 6;
  int half = lane >> 4;            // 0 or 1
  int kA = half << 4;              // A: +0 or +16
  int kB = half << 4;              // B: +0 or +16
  int ncol = lane & 15;
  const _Float16* arow = A + (size_t)(mt * 16 + ncol) * K;  // row = mt*16 + (lane&15)
  const _Float16* b0 = Bt + (size_t)(nt +  0 + ncol) * K + kB;
  const _Float16* b1 = Bt + (size_t)(nt + 16 + ncol) * K + kB;
  const _Float16* b2 = Bt + (size_t)(nt + 32 + ncol) * K + kB;
  const _Float16* b3 = Bt + (size_t)(nt + 48 + ncol) * K + kB;
  v8f c0 = {}, c1 = {}, c2 = {}, c3 = {};
  for (int k0 = 0; k0 < K; k0 += 32) {
    Frag a, f0, f1, f2, f3;
    a.h[0]  = *(const v8h*)(arow + k0 + kA);
    a.h[1]  = *(const v8h*)(arow + k0 + kA + 8);
    f0.h[0] = *(const v8h*)(b0 + k0);  f0.h[1] = *(const v8h*)(b0 + k0 + 8);
    f1.h[0] = *(const v8h*)(b1 + k0);  f1.h[1] = *(const v8h*)(b1 + k0 + 8);
    f2.h[0] = *(const v8h*)(b2 + k0);  f2.h[1] = *(const v8h*)(b2 + k0 + 8);
    f3.h[0] = *(const v8h*)(b3 + k0);  f3.h[1] = *(const v8h*)(b3 + k0 + 8);
    c0 = __builtin_amdgcn_wmma_f32_16x16x32_f16(false, a.v, false, f0.v, (short)0, c0, false, false);
    c1 = __builtin_amdgcn_wmma_f32_16x16x32_f16(false, a.v, false, f1.v, (short)0, c1, false, false);
    c2 = __builtin_amdgcn_wmma_f32_16x16x32_f16(false, a.v, false, f2.v, (short)0, c2, false, false);
    c3 = __builtin_amdgcn_wmma_f32_16x16x32_f16(false, a.v, false, f3.v, (short)0, c3, false, false);
  }
  int srow = mt * 16 + (half << 3);
  for (int r = 0; r < 8; ++r) {
    float* o = C + (size_t)(srow + r) * Nout + nt + ncol;
    o[0] = c0[r]; o[16] = c1[r]; o[32] = c2[r]; o[48] = c3[r];
  }
}

// ---------------- fused NFM: nfm = 0.5*((x@F)^2 - x@F2), Nout = 64
__global__ __launch_bounds__(256) void k_nfm_wmma(
    const _Float16* __restrict__ A, const _Float16* __restrict__ Ft,
    const _Float16* __restrict__ F2t, float* __restrict__ nfm,
    int K, int totalWaves) {
  int w    = (blockIdx.x * blockDim.x + threadIdx.x) >> 5;
  int lane = threadIdx.x & 31;
  if (w >= totalWaves) return;
  int mt = w;
  int half = lane >> 4;
  int kA = half << 4;
  int kB = half << 4;
  int ncol = lane & 15;
  const _Float16* arow = A + (size_t)(mt * 16 + ncol) * K;
  const _Float16* p0 = Ft  + (size_t)( 0 + ncol) * K + kB;
  const _Float16* p1 = Ft  + (size_t)(16 + ncol) * K + kB;
  const _Float16* p2 = Ft  + (size_t)(32 + ncol) * K + kB;
  const _Float16* p3 = Ft  + (size_t)(48 + ncol) * K + kB;
  const _Float16* q0 = F2t + (size_t)( 0 + ncol) * K + kB;
  const _Float16* q1 = F2t + (size_t)(16 + ncol) * K + kB;
  const _Float16* q2 = F2t + (size_t)(32 + ncol) * K + kB;
  const _Float16* q3 = F2t + (size_t)(48 + ncol) * K + kB;
  v8f s0 = {}, s1 = {}, s2 = {}, s3 = {};
  v8f t0 = {}, t1 = {}, t2 = {}, t3 = {};
  for (int k0 = 0; k0 < K; k0 += 32) {
    Frag a, f;
    a.h[0] = *(const v8h*)(arow + k0 + kA);
    a.h[1] = *(const v8h*)(arow + k0 + kA + 8);
    f.h[0] = *(const v8h*)(p0 + k0); f.h[1] = *(const v8h*)(p0 + k0 + 8);
    s0 = __builtin_amdgcn_wmma_f32_16x16x32_f16(false, a.v, false, f.v, (short)0, s0, false, false);
    f.h[0] = *(const v8h*)(p1 + k0); f.h[1] = *(const v8h*)(p1 + k0 + 8);
    s1 = __builtin_amdgcn_wmma_f32_16x16x32_f16(false, a.v, false, f.v, (short)0, s1, false, false);
    f.h[0] = *(const v8h*)(p2 + k0); f.h[1] = *(const v8h*)(p2 + k0 + 8);
    s2 = __builtin_amdgcn_wmma_f32_16x16x32_f16(false, a.v, false, f.v, (short)0, s2, false, false);
    f.h[0] = *(const v8h*)(p3 + k0); f.h[1] = *(const v8h*)(p3 + k0 + 8);
    s3 = __builtin_amdgcn_wmma_f32_16x16x32_f16(false, a.v, false, f.v, (short)0, s3, false, false);
    f.h[0] = *(const v8h*)(q0 + k0); f.h[1] = *(const v8h*)(q0 + k0 + 8);
    t0 = __builtin_amdgcn_wmma_f32_16x16x32_f16(false, a.v, false, f.v, (short)0, t0, false, false);
    f.h[0] = *(const v8h*)(q1 + k0); f.h[1] = *(const v8h*)(q1 + k0 + 8);
    t1 = __builtin_amdgcn_wmma_f32_16x16x32_f16(false, a.v, false, f.v, (short)0, t1, false, false);
    f.h[0] = *(const v8h*)(q2 + k0); f.h[1] = *(const v8h*)(q2 + k0 + 8);
    t2 = __builtin_amdgcn_wmma_f32_16x16x32_f16(false, a.v, false, f.v, (short)0, t2, false, false);
    f.h[0] = *(const v8h*)(q3 + k0); f.h[1] = *(const v8h*)(q3 + k0 + 8);
    t3 = __builtin_amdgcn_wmma_f32_16x16x32_f16(false, a.v, false, f.v, (short)0, t3, false, false);
  }
  int srow = mt * 16 + (half << 3);
  for (int r = 0; r < 8; ++r) {
    float* o = nfm + (size_t)(srow + r) * 64 + ncol;
    o[0]  = 0.5f * (s0[r] * s0[r] - t0[r]);
    o[16] = 0.5f * (s1[r] * s1[r] - t1[r]);
    o[32] = 0.5f * (s2[r] * s2[r] - t2[r]);
    o[48] = 0.5f * (s3[r] * s3[r] - t3[r]);
  }
}

// ---------------- f1/f2 GEMV: one wave per node ----------------
__global__ void k_f12(const float* __restrict__ Hw, const float* __restrict__ va,
                      const float* __restrict__ vb, float* __restrict__ f1,
                      float* __restrict__ f2, int n, int d) {
  int w    = (blockIdx.x * blockDim.x + threadIdx.x) >> 5;
  int lane = threadIdx.x & 31;
  if (w >= n) return;
  int per = d >> 5;
  const float* row = Hw + (size_t)w * d + lane * per;
  const float* a = va + lane * per;
  const float* b = vb + lane * per;
  float s1 = 0.f, s2 = 0.f;
  for (int i = 0; i < per; ++i) { float h = row[i]; s1 += h * a[i]; s2 += h * b[i]; }
  for (int off = 16; off > 0; off >>= 1) {
    s1 += __shfl_down(s1, off);
    s2 += __shfl_down(s2, off);
  }
  if (lane == 0) { f1[w] = s1; f2[w] = s2; }
}

// ---------------- edge pass 1: s = sigmoid(a*(f1[src]+f2[dst])); segment max
__global__ void k_edge_sig_max(const float* __restrict__ aval, const int* __restrict__ src,
                               const int* __restrict__ dst, const float* __restrict__ f1,
                               const float* __restrict__ f2, float* __restrict__ sbuf,
                               float* __restrict__ m, int E) {
  int i = blockIdx.x * blockDim.x + threadIdx.x;
  if (i >= E) return;
  int s_ = src[i], d_ = dst[i];
  float l = aval[i] * (f1[s_] + f2[d_]);
  float sg = 1.0f / (1.0f + __expf(-l));   // in (0,1) -> positive-float int max is exact
  sbuf[i] = sg;
  atomicMax((int*)(m + s_), __float_as_int(sg));
}

// ---------------- edge pass 2: e = exp(s - m[src]); segment sum
__global__ void k_edge_exp_sum(const int* __restrict__ src, const float* __restrict__ m,
                               float* __restrict__ sbuf, float* __restrict__ denom, int E) {
  int i = blockIdx.x * blockDim.x + threadIdx.x;
  if (i >= E) return;
  int s_ = src[i];
  float e = __expf(sbuf[i] - m[s_]);
  sbuf[i] = e;
  atomicAdd(denom + s_, e);
}

// ---------------- aggregation: out[src] += (e/denom[src]) * Hw[dst]
// Persistent waves grid-stride over edges; next edge's indices are loaded early
// and its Hw row is software-prefetched (global_prefetch_b8) while the current
// edge's gather+atomics are in flight. Hw is L2-resident (<= 102 MB < 192 MB L2).
__global__ __launch_bounds__(256) void k_agg(
    const float* __restrict__ Hw, const float* __restrict__ ebuf,
    const float* __restrict__ denom, const int* __restrict__ src,
    const int* __restrict__ dst, float* __restrict__ out,
    int E, int D, int totalWaves) {
  int w    = (blockIdx.x * blockDim.x + threadIdx.x) >> 5;
  int lane = threadIdx.x & 31;
  if (w >= totalWaves) return;
  int nf4 = D >> 7;  // float4s per lane (2 for D=256, 1 for D=128)

  // software-pipelined index fetch
  int s_ = 0, d_ = 0; float att = 0.f;
  if (w < E && lane == 0) { s_ = src[w]; d_ = dst[w]; att = ebuf[w] / denom[s_]; }

  for (int e = w; e < E; e += totalWaves) {
    s_  = __shfl(s_, 0);
    d_  = __shfl(d_, 0);
    att = __shfl(att, 0);

    // prefetch next edge's gather row while this edge drains
    int en = e + totalWaves;
    int dnext = 0;
    if (en < E && lane == 0) dnext = dst[en];
    dnext = __shfl(dnext, 0);
    if (en < E)
      __builtin_prefetch(Hw + (size_t)dnext * D + lane * (nf4 * 4), 0, 3);

    const float4* hin = (const float4*)(Hw + (size_t)d_ * D);
    float* op = out + (size_t)s_ * D;
    #pragma unroll 2
    for (int i = 0; i < nf4; ++i) {
      int e4 = lane * nf4 + i;
      float4 v = hin[e4];
      int base = e4 * 4;
      atomicAdd(op + base + 0, att * v.x);
      atomicAdd(op + base + 1, att * v.y);
      atomicAdd(op + base + 2, att * v.z);
      atomicAdd(op + base + 3, att * v.w);
    }

    // load next edge's scalars (lane 0) for the following iteration
    if (en < E && lane == 0) {
      s_ = src[en]; d_ = dnext; att = ebuf[en] / denom[s_];
    }
  }
}

// ---------------- final: out = [H2 | nfm] @ projW + b (projW staged in LDS)
__global__ __launch_bounds__(256) void k_final_proj(
    const float* __restrict__ H2, const float* __restrict__ nfm,
    const float* __restrict__ projW, const float* __restrict__ projb,
    float* __restrict__ out, int n) {
  __shared__ float sW[192 * 64];
  __shared__ float sB[64];
  __shared__ float rows[4][192];
  int tid = threadIdx.x;
  for (int i = tid; i < 192 * 64; i += 256) sW[i] = projW[i];
  if (tid < 64) sB[tid] = projb[tid];
  __syncthreads();
  int groups = n >> 2;
  int nl = tid >> 6, c = tid & 63;
  for (int g = blockIdx.x; g < groups; g += gridDim.x) {
    int n0 = g << 2;
    for (int i = tid; i < 4 * 192; i += 256) {
      int r = i / 192, k = i - r * 192;
      rows[r][k] = (k < 128) ? H2[(size_t)(n0 + r) * 128 + k]
                             : nfm[(size_t)(n0 + r) * 64 + (k - 128)];
    }
    __syncthreads();
    float acc = sB[c];
    for (int k = 0; k < 192; ++k) acc += rows[nl][k] * sW[k * 64 + c];
    out[(size_t)(n0 + nl) * 64 + c] = acc;
    __syncthreads();
  }
}

// ---------------- host side ----------------
static inline char* ws_take(char*& p, size_t bytes) {
  char* r = p;
  p += (bytes + 255) & ~(size_t)255;
  return r;
}

extern "C" void kernel_launch(void* const* d_in, const int* in_sizes, int n_in,
                              void* d_out, int out_size, void* d_ws, size_t ws_size,
                              hipStream_t stream) {
  const float* x     = (const float*)d_in[0];
  const float* aval  = (const float*)d_in[1];
  const float* W0    = (const float*)d_in[2];
  const float* W1    = (const float*)d_in[3];
  const float* v0_0  = (const float*)d_in[4];
  const float* v0_1  = (const float*)d_in[5];
  const float* v1_0  = (const float*)d_in[6];
  const float* v1_1  = (const float*)d_in[7];
  const float* fm    = (const float*)d_in[8];
  const float* projW = (const float*)d_in[9];
  const float* projb = (const float*)d_in[10];
  const int*   esrc  = (const int*)d_in[11];
  const int*   edst  = (const int*)d_in[12];
  float* out = (float*)d_out;

  const int D_IN = 512, D1 = 256, D2 = 128, F = 64;
  const int N = in_sizes[0] / D_IN;     // 100000
  const int E = in_sizes[1];            // 3200000

  char* p = (char*)d_ws;
  _Float16* x_h   = (_Float16*)ws_take(p, (size_t)N * D_IN * 2);
  _Float16* W0t   = (_Float16*)ws_take(p, (size_t)D_IN * D1 * 2);
  _Float16* W1t   = (_Float16*)ws_take(p, (size_t)D1 * D2 * 2);
  _Float16* fmT   = (_Float16*)ws_take(p, (size_t)D_IN * F * 2);
  _Float16* fm2T  = (_Float16*)ws_take(p, (size_t)D_IN * F * 2);
  float*    Hw0   = (float*)ws_take(p, (size_t)N * D1 * 4);   // also reused as Hw1
  float*    H1    = (float*)ws_take(p, (size_t)N * D1 * 4);   // also reused as H2
  _Float16* H1h   = (_Float16*)ws_take(p, (size_t)N * D1 * 2);
  float*    f1    = (float*)ws_take(p, (size_t)N * 4);
  float*    f2    = (float*)ws_take(p, (size_t)N * 4);
  float*    m     = (float*)ws_take(p, (size_t)N * 4);
  float*    dn    = (float*)ws_take(p, (size_t)N * 4);
  float*    sbuf  = (float*)ws_take(p, (size_t)E * 4);
  float*    nfm   = (float*)ws_take(p, (size_t)N * F * 4);
  float*    Hw1   = Hw0;   // alias: Hw0 dead after layer-1 aggregate
  float*    H2    = H1;    // alias: H1 (f32) dead after conversion to H1h
  (void)ws_size; (void)n_in; (void)out_size;

  const int aggWaves  = 65536;           // persistent waves for aggregation
  const int aggBlocks = aggWaves * 32 / 256;

  // ---- prep: f16 conversions + transposed weights
  k_cvt_f16<<<(N * D_IN + 255) / 256, 256, 0, stream>>>(x, x_h, N * D_IN);
  k_transpose_f16<<<(D_IN * D1 + 255) / 256, 256, 0, stream>>>(W0, W0t, D_IN, D1, 0);
  k_transpose_f16<<<(D1 * D2 + 255) / 256, 256, 0, stream>>>(W1, W1t, D1, D2, 0);
  k_transpose_f16<<<(D_IN * F + 255) / 256, 256, 0, stream>>>(fm, fmT, D_IN, F, 0);
  k_transpose_f16<<<(D_IN * F + 255) / 256, 256, 0, stream>>>(fm, fm2T, D_IN, F, 1);

  // ---- layer 1: Hw0 = x @ W0  (WMMA)
  {
    int waves = (N / 16) * (D1 / 64);
    k_gemm_wmma<<<(waves * 32 + 255) / 256, 256, 0, stream>>>(x_h, W0t, Hw0, D_IN, D1, waves);
  }
  k_f12<<<(N * 32 + 255) / 256, 256, 0, stream>>>(Hw0, v0_0, v0_1, f1, f2, N, D1);
  hipMemsetAsync(m, 0, (size_t)N * 4, stream);
  hipMemsetAsync(dn, 0, (size_t)N * 4, stream);
  hipMemsetAsync(H1, 0, (size_t)N * D1 * 4, stream);
  k_edge_sig_max<<<(E + 255) / 256, 256, 0, stream>>>(aval, esrc, edst, f1, f2, sbuf, m, E);
  k_edge_exp_sum<<<(E + 255) / 256, 256, 0, stream>>>(esrc, m, sbuf, dn, E);
  k_agg<<<aggBlocks, 256, 0, stream>>>(Hw0, sbuf, dn, esrc, edst, H1, E, D1, aggWaves);
  k_cvt_f16<<<(N * D1 + 255) / 256, 256, 0, stream>>>(H1, H1h, N * D1);

  // ---- layer 2: Hw1 = H1 @ W1  (WMMA)
  {
    int waves = (N / 16) * (D2 / 64);
    k_gemm_wmma<<<(waves * 32 + 255) / 256, 256, 0, stream>>>(H1h, W1t, Hw1, D1, D2, waves);
  }
  k_f12<<<(N * 32 + 255) / 256, 256, 0, stream>>>(Hw1, v1_0, v1_1, f1, f2, N, D2);
  hipMemsetAsync(m, 0, (size_t)N * 4, stream);
  hipMemsetAsync(dn, 0, (size_t)N * 4, stream);
  hipMemsetAsync(H2, 0, (size_t)N * D2 * 4, stream);
  k_edge_sig_max<<<(E + 255) / 256, 256, 0, stream>>>(aval, esrc, edst, f1, f2, sbuf, m, E);
  k_edge_exp_sum<<<(E + 255) / 256, 256, 0, stream>>>(esrc, m, sbuf, dn, E);
  k_agg<<<aggBlocks, 256, 0, stream>>>(Hw1, sbuf, dn, esrc, edst, H2, E, D2, aggWaves);

  // ---- NFM (fused two-GEMM WMMA)
  {
    int waves = N / 16;
    k_nfm_wmma<<<(waves * 32 + 255) / 256, 256, 0, stream>>>(x_h, fmT, fm2T, nfm, D_IN, waves);
  }

  // ---- final projection
  k_final_proj<<<2048, 256, 0, stream>>>(H2, nfm, projW, projb, out, N);
}
